// VLinearMarlin_50611894616845
// MI455X (gfx1250) — matrix operand order repaired and checked by reading, not compiled, not run on hardware
//
#include <hip/hip_runtime.h>

// ---------------------------------------------------------------------------
// Marlin-style int4-grouped dequant GEMM for gfx1250 (CDNA5, wave32, WMMA).
//   C[M,N] = x[M,K] @ ((q[K,N] - 8) * scales[K/64, N])
// M=4096, K=4096, N=14336, GROUP=64.
//
// Pass 1: x fp32 -> bf16 (into d_ws) once (~4us at 23.3 TB/s).
// Pass 2: 256x128x64 tiles, 512 threads (16 waves), double-buffered LDS:
//   - A tile t+1: global_load_async_to_lds_b128 (ASYNCcnt) issued pre-compute
//   - B tile t+1: int4 global loads into regs pre-compute; dequant
//     w = cvt_f32(q)*s + (-8s) packed bf16x2 -> LDS post-compute
//   - 16x16x32 bf16 WMMA, f32 accum; 16 WMMA / wave / k-tile; 1 barrier/tile
// ---------------------------------------------------------------------------

typedef __attribute__((ext_vector_type(16))) __bf16 v16bf;
typedef __attribute__((ext_vector_type(8)))  float  v8f;

namespace {
constexpr int kM = 4096;
constexpr int kK = 4096;
constexpr int kN = 14336;
constexpr int kGroup = 64;

constexpr int BM = 256;
constexpr int BN = 128;
constexpr int BK = 64;          // == kGroup: one scale row per k-tile
constexpr int LDA = BK + 8;     // 72 bf16 stride -> 144 B (16B aligned)
constexpr int LDB = BK + 8;
constexpr int NT  = 512;        // 16 waves
constexpr int KTILES = kK / BK; // 64
}

union FragBF {
  uint4 u[2];
  v16bf v;
};

__device__ inline unsigned pk_bf16(float a, float b) {
  __bf16 x = (__bf16)a;
  __bf16 y = (__bf16)b;
  return (unsigned)__builtin_bit_cast(unsigned short, x) |
         ((unsigned)__builtin_bit_cast(unsigned short, y) << 16);
}

// ---- pass 1: convert x fp32 -> bf16 row-major into workspace ----
__global__ __launch_bounds__(256) void
cvt_x_bf16(const float* __restrict__ x, unsigned short* __restrict__ xbf) {
  const size_t i = ((size_t)blockIdx.x * 256 + threadIdx.x) * 8;
  const float4 a = *(const float4*)(x + i);
  const float4 b = *(const float4*)(x + i + 4);
  uint4 o;
  o.x = pk_bf16(a.x, a.y);
  o.y = pk_bf16(a.z, a.w);
  o.z = pk_bf16(b.x, b.y);
  o.w = pk_bf16(b.z, b.w);
  *(uint4*)(xbf + i) = o;
}

// ---- pass 2 (main): double-buffered, async-A, pipelined-B ----
__global__ __launch_bounds__(NT) void
marlin_pipe(const unsigned short* __restrict__ xbf,
            const int*            __restrict__ qw,
            const float*          __restrict__ scales,
            float*                __restrict__ out) {
  __shared__ __bf16 As[2][BM * LDA];   // 73,728 B
  __shared__ __bf16 Bs[2][BN * LDB];   // 36,864 B  (total 110,592 <= 320KB/WGP)

  const int tid  = threadIdx.x;
  const int lane = tid & 31;
  const int wave = tid >> 5;        // 0..15
  const int wm   = wave >> 2;       // 0..3 : 64-row slice
  const int wn   = wave & 3;        // 0..3 : 32-col slice
  const int l15  = lane & 15;
  const int kh   = lane >> 4;       // K-half selector per WMMA layout

  const int row0 = blockIdx.y * BM;
  const int col0 = blockIdx.x * BN;

  // B staging: fixed 4 columns, 4 consecutive k-rows per thread
  const int bc  = (tid & 31) << 2;  // column in tile  (0..124)
  const int bk0 = (tid >> 5) << 2;  // first k-row     (0..60)

  v8f zero;
#pragma unroll
  for (int i = 0; i < 8; ++i) zero[i] = 0.0f;

  v8f acc[4][2];
#pragma unroll
  for (int mf = 0; mf < 4; ++mf)
#pragma unroll
    for (int nf = 0; nf < 2; ++nf) acc[mf][nf] = zero;

  // pipeline registers for B
  int4   qreg[4];
  float4 sreg;

  auto issueA = [&](int k0, int buf) {
#pragma unroll
    for (int i = 0; i < 4; ++i) {
      const int idx = tid + i * NT;       // 0..2047 8-elem chunks
      const int r   = idx >> 3;
      const int cu  = (idx & 7) << 3;
      const unsigned short* gp = xbf + (size_t)(row0 + r) * kK + (k0 + cu);
      const unsigned lds = (unsigned)(size_t)(&As[buf][r * LDA + cu]);
      asm volatile("global_load_async_to_lds_b128 %0, %1, off"
                   :: "v"(lds), "v"(gp) : "memory");
    }
  };

  auto loadB = [&](int k0) {
    const int g = k0 >> 6;                // BK == kGroup == 64
    sreg = *(const float4*)(scales + (size_t)g * kN + col0 + bc);
#pragma unroll
    for (int i = 0; i < 4; ++i)
      qreg[i] = *(const int4*)(qw + (size_t)(k0 + bk0 + i) * kN + col0 + bc);
  };

  auto storeB = [&](int buf) {
    const float4 s4 = sreg;
    const float4 n8 = make_float4(-8.0f * s4.x, -8.0f * s4.y,
                                  -8.0f * s4.z, -8.0f * s4.w);
#pragma unroll
    for (int i = 0; i < 4; i += 2) {
      const int4 qa = qreg[i];
      const int4 qb = qreg[i + 1];
      *(unsigned*)(&Bs[buf][(bc + 0) * LDB + bk0 + i]) =
          pk_bf16((float)qa.x * s4.x + n8.x, (float)qb.x * s4.x + n8.x);
      *(unsigned*)(&Bs[buf][(bc + 1) * LDB + bk0 + i]) =
          pk_bf16((float)qa.y * s4.y + n8.y, (float)qb.y * s4.y + n8.y);
      *(unsigned*)(&Bs[buf][(bc + 2) * LDB + bk0 + i]) =
          pk_bf16((float)qa.z * s4.z + n8.z, (float)qb.z * s4.z + n8.z);
      *(unsigned*)(&Bs[buf][(bc + 3) * LDB + bk0 + i]) =
          pk_bf16((float)qa.w * s4.w + n8.w, (float)qb.w * s4.w + n8.w);
    }
  };

  auto compute = [&](int buf) {
#pragma unroll
    for (int kk = 0; kk < BK; kk += 32) {
      FragBF a[4], b[2];
#pragma unroll
      for (int mf = 0; mf < 4; ++mf) {
        const __bf16* p =
            &As[buf][(wm * 64 + mf * 16 + l15) * LDA + kk + kh * 8];
        a[mf].u[0] = *(const uint4*)(p);
        a[mf].u[1] = *(const uint4*)(p + 16);
      }
#pragma unroll
      for (int nf = 0; nf < 2; ++nf) {
        const __bf16* p =
            &Bs[buf][(wn * 32 + nf * 16 + l15) * LDB + kk + kh * 8];
        b[nf].u[0] = *(const uint4*)(p);
        b[nf].u[1] = *(const uint4*)(p + 16);
      }
#pragma unroll
      for (int mf = 0; mf < 4; ++mf)
#pragma unroll
        for (int nf = 0; nf < 2; ++nf)
          acc[mf][nf] = __builtin_amdgcn_wmma_f32_16x16x32_bf16(
              false, a[mf].v, false, b[nf].v,
              (short)0, acc[mf][nf], false, false);
    }
  };

  // ---- prologue: stage tile 0 into buffer 0 ----
  issueA(0, 0);
  loadB(0);
  storeB(0);
  asm volatile("s_wait_asynccnt 0x0" ::: "memory");
  __syncthreads();

  // ---- pipelined main loop: one barrier per tile ----
  for (int t = 0; t < KTILES; ++t) {
    const int cur  = t & 1;
    const int nxt  = cur ^ 1;
    const bool more = (t + 1) < KTILES;
    if (more) {
      issueA((t + 1) * BK, nxt);   // async DMA overlaps compute below
      loadB((t + 1) * BK);         // global loads issue now, consumed later
    }
    compute(cur);
    if (more) {
      storeB(nxt);                 // dequant + LDS stores after compute
      asm volatile("s_wait_asynccnt 0x0" ::: "memory");
    }
    __syncthreads();
  }

  // ---- epilogue ----
#pragma unroll
  for (int mf = 0; mf < 4; ++mf) {
    const int mbase = row0 + wm * 64 + mf * 16 + kh * 8;
#pragma unroll
    for (int nf = 0; nf < 2; ++nf) {
      const int n = col0 + wn * 32 + nf * 16 + l15;
#pragma unroll
      for (int r = 0; r < 8; ++r)
        out[(size_t)(mbase + r) * kN + n] = acc[mf][nf][r];
    }
  }
}

// ---- pass 2 (fallback, no workspace): unpipelined, inline A conversion ----
__global__ __launch_bounds__(NT) void
marlin_fallback(const float* __restrict__ x,
                const int*   __restrict__ qw,
                const float* __restrict__ scales,
                float*       __restrict__ out) {
  __shared__ __bf16 As[BM * LDA];
  __shared__ __bf16 Bs[BN * LDB];

  const int tid  = threadIdx.x;
  const int lane = tid & 31;
  const int wave = tid >> 5;
  const int wm   = wave >> 2;
  const int wn   = wave & 3;
  const int l15  = lane & 15;
  const int kh   = lane >> 4;

  const int row0 = blockIdx.y * BM;
  const int col0 = blockIdx.x * BN;
  const int bc   = (tid & 31) << 2;
  const int bk0  = (tid >> 5) << 2;

  v8f zero;
#pragma unroll
  for (int i = 0; i < 8; ++i) zero[i] = 0.0f;

  v8f acc[4][2];
#pragma unroll
  for (int mf = 0; mf < 4; ++mf)
#pragma unroll
    for (int nf = 0; nf < 2; ++nf) acc[mf][nf] = zero;

  for (int k0 = 0; k0 < kK; k0 += BK) {
#pragma unroll
    for (int i = 0; i < 8; ++i) {
      const int idx = tid + i * NT;
      const int r   = idx >> 4;
      const int c   = (idx & 15) << 2;
      const float4 v =
          *(const float4*)(x + (size_t)(row0 + r) * kK + (k0 + c));
      uint2 p;
      p.x = pk_bf16(v.x, v.y);
      p.y = pk_bf16(v.z, v.w);
      *(uint2*)(&As[r * LDA + c]) = p;
    }
    const int g = k0 / kGroup;
    const float4 s4 = *(const float4*)(scales + (size_t)g * kN + col0 + bc);
    const float4 n8 = make_float4(-8.0f * s4.x, -8.0f * s4.y,
                                  -8.0f * s4.z, -8.0f * s4.w);
#pragma unroll
    for (int i = 0; i < 4; i += 2) {
      const int4 qa =
          *(const int4*)(qw + (size_t)(k0 + bk0 + i)     * kN + col0 + bc);
      const int4 qb =
          *(const int4*)(qw + (size_t)(k0 + bk0 + i + 1) * kN + col0 + bc);
      *(unsigned*)(&Bs[(bc + 0) * LDB + bk0 + i]) =
          pk_bf16((float)qa.x * s4.x + n8.x, (float)qb.x * s4.x + n8.x);
      *(unsigned*)(&Bs[(bc + 1) * LDB + bk0 + i]) =
          pk_bf16((float)qa.y * s4.y + n8.y, (float)qb.y * s4.y + n8.y);
      *(unsigned*)(&Bs[(bc + 2) * LDB + bk0 + i]) =
          pk_bf16((float)qa.z * s4.z + n8.z, (float)qb.z * s4.z + n8.z);
      *(unsigned*)(&Bs[(bc + 3) * LDB + bk0 + i]) =
          pk_bf16((float)qa.w * s4.w + n8.w, (float)qb.w * s4.w + n8.w);
    }
    __syncthreads();

#pragma unroll
    for (int kk = 0; kk < BK; kk += 32) {
      FragBF a[4], b[2];
#pragma unroll
      for (int mf = 0; mf < 4; ++mf) {
        const __bf16* p = &As[(wm * 64 + mf * 16 + l15) * LDA + kk + kh * 8];
        a[mf].u[0] = *(const uint4*)(p);
        a[mf].u[1] = *(const uint4*)(p + 16);
      }
#pragma unroll
      for (int nf = 0; nf < 2; ++nf) {
        const __bf16* p = &Bs[(wn * 32 + nf * 16 + l15) * LDB + kk + kh * 8];
        b[nf].u[0] = *(const uint4*)(p);
        b[nf].u[1] = *(const uint4*)(p + 16);
      }
#pragma unroll
      for (int mf = 0; mf < 4; ++mf)
#pragma unroll
        for (int nf = 0; nf < 2; ++nf)
          acc[mf][nf] = __builtin_amdgcn_wmma_f32_16x16x32_bf16(
              false, a[mf].v, false, b[nf].v,
              (short)0, acc[mf][nf], false, false);
    }
    __syncthreads();
  }

#pragma unroll
  for (int mf = 0; mf < 4; ++mf) {
    const int mbase = row0 + wm * 64 + mf * 16 + kh * 8;
#pragma unroll
    for (int nf = 0; nf < 2; ++nf) {
      const int n = col0 + wn * 32 + nf * 16 + l15;
#pragma unroll
      for (int r = 0; r < 8; ++r)
        out[(size_t)(mbase + r) * kN + n] = acc[mf][nf][r];
    }
  }
}

extern "C" void kernel_launch(void* const* d_in, const int* in_sizes, int n_in,
                              void* d_out, int out_size, void* d_ws,
                              size_t ws_size, hipStream_t stream) {
  const float* x      = (const float*)d_in[0];
  const int*   qw     = (const int*)d_in[1];
  const float* scales = (const float*)d_in[2];
  float*       out    = (float*)d_out;

  const size_t xbf_bytes = (size_t)kM * kK * sizeof(unsigned short);
  const dim3 grid(kN / BN, kM / BM);   // 112 x 16

  if (ws_size >= xbf_bytes) {
    unsigned short* xbf = (unsigned short*)d_ws;
    cvt_x_bf16<<<dim3((unsigned)((size_t)kM * kK / (256 * 8))), dim3(256), 0,
                 stream>>>(x, xbf);
    marlin_pipe<<<grid, dim3(NT), 0, stream>>>(xbf, qw, scales, out);
  } else {
    marlin_fallback<<<grid, dim3(NT), 0, stream>>>(x, qw, scales, out);
  }
}
